// VectorQuantizerEMA_86182813762429
// MI455X (gfx1250) — matrix operand order repaired and checked by reading, not compiled, not run on hardware
//
#include <hip/hip_runtime.h>

typedef __bf16 bf16_t;
typedef __attribute__((ext_vector_type(16))) __bf16 bf16x16;
typedef __attribute__((ext_vector_type(8)))  __bf16 bf16x8;
typedef __attribute__((ext_vector_type(8)))  float  f32x8;
typedef __attribute__((ext_vector_type(4))) unsigned int u32x4;
typedef __attribute__((ext_vector_type(8))) int         i32x8;
typedef __attribute__((ext_vector_type(4))) int         i32x4;

#define KCODES 1024
#define DDIM   64
#define TDIM   2048
#define BDIM   32
#define NVEC   (BDIM*TDIM)        // 65536 flat vectors
#define ZELEMS (NVEC*DDIM)        // 4194304 elements of z

#define CHUNK_CODES 128                        // codes staged per LDS chunk
#define NCHUNK      (KCODES / CHUNK_CODES)     // 8
#define CHUNK_ELEMS (CHUNK_CODES * DDIM)       // 8192 bf16 = 16 KB

// ---- workspace layout (bytes) ----
#define WS_FLATBF 0                                    // [NVEC*DDIM] bf16  (8 MB)
#define WS_EMBBF  (ZELEMS*2)                           // [KCODES*DDIM] bf16
#define WS_EN2    (WS_EMBBF + KCODES*DDIM*2)           // [KCODES] f32
#define WS_IDX    (WS_EN2 + KCODES*4)                  // [NVEC] i32
#define WS_CNT    (WS_IDX + NVEC*4)                    // [KCODES] f32
#define WS_DW     (WS_CNT + KCODES*4)                  // [KCODES*DDIM] f32
#define WS_LOSS   (WS_DW + KCODES*DDIM*4)              // [1] f32

// ------------------------------------------------------------------
// 0) zero the accumulators (counts + dw + loss are contiguous in ws)
// ------------------------------------------------------------------
__global__ void vq_zero(float* __restrict__ p, int n) {
    int i = blockIdx.x * 256 + threadIdx.x;
    if (i < n) p[i] = 0.0f;
}

// ------------------------------------------------------------------
// 1) z [B,D,T] f32 -> flat [N=B*T, D] bf16 via LDS tile transpose.
//    64(d) x 64(t) tile per block; both global read and write coalesced.
// ------------------------------------------------------------------
__global__ __launch_bounds__(256)
void vq_prep_z(const float* __restrict__ z, bf16_t* __restrict__ flatbf) {
    __shared__ bf16_t s[64][66];                 // pad to spread LDS banks
    const int b  = blockIdx.x >> 5;              // batch index
    const int tt = (blockIdx.x & 31) << 6;       // t-tile base
    const int tid = threadIdx.x;
    #pragma unroll
    for (int l = 0; l < 16; ++l) {
        int e = l * 256 + tid;
        int d = e >> 6, x = e & 63;              // coalesced along t
        s[d][x] = (bf16_t)z[(((size_t)b << 6) + d) * TDIM + tt + x];
    }
    __syncthreads();
    #pragma unroll
    for (int l = 0; l < 16; ++l) {
        int e = l * 256 + tid;
        int row = e >> 6, c = e & 63;            // coalesced along d
        flatbf[(((size_t)b << 11) + tt + row) * DDIM + c] = s[c][row];
    }
}

// ------------------------------------------------------------------
// 2) embedding -> bf16 copy + per-code squared norm (f32)
// ------------------------------------------------------------------
__global__ __launch_bounds__(256)
void vq_prep_emb(const float* __restrict__ emb, bf16_t* __restrict__ embbf,
                 float* __restrict__ en2) {
    int k = blockIdx.x * 256 + threadIdx.x;
    if (k >= KCODES) return;
    float s = 0.0f;
    #pragma unroll 8
    for (int d = 0; d < DDIM; ++d) {
        float v = emb[k * DDIM + d];
        s += v * v;
        embbf[k * DDIM + d] = (bf16_t)v;
    }
    en2[k] = s;
}

// ------------------------------------------------------------------
// TDM: 1-D async copy global -> LDS (nbytes must be dword-multiple).
// D# per CDNA5 ISA: count=1, type=2,
// data_size=4B, tensor_dim0 = tile_dim0 = ndwords, tensor_dim1 = 1.
// This toolchain exposes the 6-arg builtin:
//   (u32x4 g0, i32x8 g1, i32x4 g2, i32x4 g3, i32x8, i32 cpol)
// ------------------------------------------------------------------
__device__ __forceinline__ void tdm_load_chunk(const bf16_t* gsrc,
                                               unsigned lds_off,
                                               unsigned nbytes) {
    unsigned ndw = nbytes >> 2;
    unsigned long long ga = (unsigned long long)(size_t)gsrc;
    u32x4 g0;
    g0[0] = 1u;                                        // count=1 (user D#)
    g0[1] = lds_off;                                   // lds_addr (bytes)
    g0[2] = (unsigned)(ga & 0xFFFFFFFFu);              // global_addr[31:0]
    g0[3] = (unsigned)((ga >> 32) & 0x1FFFFFFu)        // global_addr[56:32]
          | (2u << 30);                                // type=2 ("image")
    i32x8 g1;
    g1[0] = (int)(2u << 16);                           // data_size=2 -> 4B
    g1[1] = (int)((ndw & 0xFFFFu) << 16);              // tensor_dim0[15:0]
    g1[2] = (int)((ndw >> 16) | (1u << 16));           // dim0[31:16] | dim1 lo=1
    g1[3] = (int)((ndw & 0xFFFFu) << 16);              // dim1 hi=0 | tile_dim0
    g1[4] = 0;                                         // tile_dim1/2 unused
    g1[5] = (int)ndw;                                  // tensor_dim0_stride lo
    g1[6] = 0;
    g1[7] = 0;
    i32x4 zero4 = {0, 0, 0, 0};
    i32x8 zero8 = {0, 0, 0, 0, 0, 0, 0, 0};
    __builtin_amdgcn_tensor_load_to_lds(g0, g1, zero4, zero4, zero8, 0);
}

// ------------------------------------------------------------------
// 3) argmin over codes via v_wmma_f32_16x16x32_bf16.
//    8 waves/block, 2 row tiles per wave (32 rows). Codebook streamed
//    through LDS in 16KB chunks, double-buffered, by the TDM engine;
//    WMMAs consume one buffer while TDM fills the other.
//    dist' = ||e||^2 - 2*(flat.e)  (||z||^2 is row-constant -> dropped)
// ------------------------------------------------------------------
__global__ __launch_bounds__(256)
void vq_argmin_wmma(const bf16_t* __restrict__ flatbf,
                    const bf16_t* __restrict__ embbf,
                    const float*  __restrict__ en2,
                    int* __restrict__ idx_out) {
    __shared__ bf16_t sEmb[2][CHUNK_ELEMS];      // 2 x 16 KB
    const int lane = threadIdx.x & 31;
    const int wv   = threadIdx.x >> 5;
    const int hlf  = lane >> 4;
    const int j    = lane & 15;
    const int tile0 = blockIdx.x * 16 + wv * 2;  // first of 2 row tiles

    // kick off chunk 0 while we load A
    if (wv == 0)
        tdm_load_chunk(embbf, (unsigned)(size_t)&sEmb[0][0], CHUNK_ELEMS * 2);

    // A operands: contiguous K layout (lanes 0-15: K 0-15,
    // lanes 16-31: K 16-31), rows m = lane&15.
    bf16x16 a0[2], a1[2];
    #pragma unroll
    for (int rt = 0; rt < 2; ++rt) {
        const bf16_t* arow = flatbf + (size_t)((tile0 + rt) * 16 + j) * DDIM;
        bf16x8 g0 = *(const bf16x8*)(arow +      hlf * 16);
        bf16x8 g1 = *(const bf16x8*)(arow +  8 + hlf * 16);
        bf16x8 g2 = *(const bf16x8*)(arow + 32 + hlf * 16);
        bf16x8 g3 = *(const bf16x8*)(arow + 40 + hlf * 16);
        #pragma unroll
        for (int i = 0; i < 8; ++i) {
            a0[rt][i] = g0[i]; a0[rt][8 + i] = g1[i];   // K = 0..31
            a1[rt][i] = g2[i]; a1[rt][8 + i] = g3[i];   // K = 32..63
        }
    }

    float minv[2][8]; int mini[2][8];
    #pragma unroll
    for (int rt = 0; rt < 2; ++rt)
        #pragma unroll
        for (int r = 0; r < 8; ++r) { minv[rt][r] = 3.4e38f; mini[rt][r] = 0; }

    for (int ch = 0; ch < NCHUNK; ++ch) {
        const int buf = ch & 1;
        if (wv == 0) __builtin_amdgcn_s_wait_tensorcnt(0);
        __syncthreads();                          // chunk ch visible to all
        if (wv == 0 && ch + 1 < NCHUNK)           // prefetch next chunk
            tdm_load_chunk(embbf + (size_t)(ch + 1) * CHUNK_ELEMS,
                           (unsigned)(size_t)&sEmb[buf ^ 1][0],
                           CHUNK_ELEMS * 2);

        for (int cc = 0; cc < CHUNK_CODES / 16; ++cc) {
            const int col = ch * CHUNK_CODES + cc * 16 + j;
            // B (64x16): lane halves split K, lane&15 = column
            const bf16_t* brow = &sEmb[buf][(cc * 16 + j) * DDIM + hlf * 16];
            bf16x16 b0 = *(const bf16x16*)(brow);        // K = 0..31
            bf16x16 b1 = *(const bf16x16*)(brow + 32);   // K = 32..63
            const float e2 = en2[col];
            #pragma unroll
            for (int rt = 0; rt < 2; ++rt) {
                f32x8 c = {};
                c = __builtin_amdgcn_wmma_f32_16x16x32_bf16(
                        false, a0[rt], false, b0, (short)0, c, false, false);
                c = __builtin_amdgcn_wmma_f32_16x16x32_bf16(
                        false, a1[rt], false, b1, (short)0, c, false, false);
                #pragma unroll
                for (int r = 0; r < 8; ++r) {     // C: VGPR r -> row hlf*8+r
                    float dist = e2 - 2.0f * c[r];
                    if (dist < minv[rt][r]) { minv[rt][r] = dist; mini[rt][r] = col; }
                }
            }
        }
        __syncthreads();                          // done reading buf
    }

    // argmin across the 16 lanes of each half (ties -> lower index)
    #pragma unroll
    for (int off = 8; off >= 1; off >>= 1) {
        #pragma unroll
        for (int rt = 0; rt < 2; ++rt)
            #pragma unroll
            for (int r = 0; r < 8; ++r) {
                float ov = __shfl_xor(minv[rt][r], off, 32);
                int   oi = __shfl_xor(mini[rt][r], off, 32);
                if (ov < minv[rt][r] ||
                    (ov == minv[rt][r] && oi < mini[rt][r])) {
                    minv[rt][r] = ov; mini[rt][r] = oi;
                }
            }
    }
    if (j == 0) {
        #pragma unroll
        for (int rt = 0; rt < 2; ++rt)
            #pragma unroll
            for (int r = 0; r < 8; ++r)
                idx_out[(tile0 + rt) * 16 + (hlf << 3) + r] = mini[rt][r];
    }
}

// ------------------------------------------------------------------
// 4) one pass in z-layout order: quantized out, codes, segment sums, loss
// ------------------------------------------------------------------
__global__ __launch_bounds__(256)
void vq_scatter(const float* __restrict__ z, const float* __restrict__ emb,
                const int* __restrict__ idx,
                float* __restrict__ out_q, float* __restrict__ out_codes,
                float* __restrict__ counts, float* __restrict__ dw,
                float* __restrict__ loss_acc) {
    __shared__ float red[256];
    int p = blockIdx.x * 256 + threadIdx.x;      // z-layout: ((b*64+d)*2048+t)
    int t = p & (TDIM - 1);
    int d = (p >> 11) & (DDIM - 1);
    int b = p >> 17;
    int n = (b << 11) + t;
    int k = idx[n];
    float zv = z[p];
    float e  = emb[k * DDIM + d];
    out_q[p] = e;                                // straight-through == quant
    atomicAdd(dw + k * DDIM + d, zv);
    if (d == 0) {
        atomicAdd(counts + k, 1.0f);
        out_codes[n] = (float)k;
    }
    float diff = e - zv;
    red[threadIdx.x] = diff * diff;
    __syncthreads();
    for (int s = 128; s > 0; s >>= 1) {
        if (threadIdx.x < s) red[threadIdx.x] += red[threadIdx.x + s];
        __syncthreads();
    }
    if (threadIdx.x == 0) atomicAdd(loss_acc, red[0]);
}

// ------------------------------------------------------------------
// 5) EMA update + Laplace smoothing + normalized embedding + loss scale
// ------------------------------------------------------------------
__global__ __launch_bounds__(1024)
void vq_finalize(const float* __restrict__ ema_cs, const float* __restrict__ ema_w,
                 const float* __restrict__ counts, const float* __restrict__ dw,
                 const float* __restrict__ loss_acc,
                 float* __restrict__ out_loss, float* __restrict__ out_emb,
                 float* __restrict__ out_cs, float* __restrict__ out_w) {
    __shared__ float red[KCODES];
    int k = threadIdx.x;
    float c = 0.99f * ema_cs[k] + 0.01f * counts[k];
    out_cs[k] = c;
    red[k] = c;
    __syncthreads();
    for (int s = 512; s > 0; s >>= 1) {
        if (k < s) red[k] += red[k + s];
        __syncthreads();
    }
    float ntot = red[0];
    float cl   = (c + 1e-5f) / (ntot + (float)KCODES * 1e-5f) * ntot;
    float inv  = 1.0f / cl;
    #pragma unroll 8
    for (int d = 0; d < DDIM; ++d) {
        float w = 0.99f * ema_w[k * DDIM + d] + 0.01f * dw[k * DDIM + d];
        out_w[k * DDIM + d]   = w;
        out_emb[k * DDIM + d] = w * inv;
    }
    if (k == 0) out_loss[0] = loss_acc[0] * (0.25f / (float)ZELEMS);
}

// ------------------------------------------------------------------
extern "C" void kernel_launch(void* const* d_in, const int* in_sizes, int n_in,
                              void* d_out, int out_size, void* d_ws, size_t ws_size,
                              hipStream_t stream) {
    const float* z   = (const float*)d_in[0];  // [32,64,2048]
    const float* emb = (const float*)d_in[1];  // [1024,64]
    const float* ecs = (const float*)d_in[2];  // [1024]
    const float* ew  = (const float*)d_in[3];  // [1024,64]

    char* ws = (char*)d_ws;
    bf16_t* flatbf = (bf16_t*)(ws + WS_FLATBF);
    bf16_t* embbf  = (bf16_t*)(ws + WS_EMBBF);
    float*  en2    = (float*)(ws + WS_EN2);
    int*    idxp   = (int*)  (ws + WS_IDX);
    float*  cnt    = (float*)(ws + WS_CNT);
    float*  dw     = (float*)(ws + WS_DW);
    float*  loss   = (float*)(ws + WS_LOSS);

    float* out       = (float*)d_out;
    float* out_q     = out;                         // 4194304
    float* out_loss  = out + ZELEMS;                // 1
    float* out_codes = out_loss + 1;                // 65536
    float* out_emb   = out_codes + NVEC;            // 65536
    float* out_cs    = out_emb + KCODES * DDIM;     // 1024
    float* out_w     = out_cs + KCODES;             // 65536

    const int nzero = KCODES + KCODES * DDIM + 1;   // counts + dw + loss
    vq_zero<<<(nzero + 255) / 256, 256, 0, stream>>>(cnt, nzero);
    vq_prep_z<<<BDIM * (TDIM / 64), 256, 0, stream>>>(z, flatbf);
    vq_prep_emb<<<(KCODES + 255) / 256, 256, 0, stream>>>(emb, embbf, en2);
    vq_argmin_wmma<<<(NVEC / 16) / 16, 256, 0, stream>>>(flatbf, embbf, en2, idxp);
    vq_scatter<<<ZELEMS / 256, 256, 0, stream>>>(z, emb, idxp, out_q, out_codes,
                                                 cnt, dw, loss);
    vq_finalize<<<1, KCODES, 0, stream>>>(ecs, ew, cnt, dw, loss,
                                          out_loss, out_emb, out_cs, out_w);
}
